// dgl_sgc_18047452578202
// MI455X (gfx1250) — matrix operand order repaired and checked by reading, not compiled, not run on hardware
//
#include <hip/hip_runtime.h>
#include <hip/hip_bf16.h>
#include <stdint.h>

typedef __attribute__((ext_vector_type(16))) __bf16 v16bf;
typedef __attribute__((ext_vector_type(8)))  float  v8f;

#define D_IN  128
#define D_HID 128
#define D_OUT 64

// Native RNE conversion: let clang emit v_cvt_*bf16_f32 on gfx1250
static __device__ __forceinline__ __bf16 f2bf(float f) { return (__bf16)f; }

// bf16 bit pattern for the weight-prep kernel's ushort store
static __device__ __forceinline__ unsigned short f2bf_bits(float f) {
  union { __bf16 b; unsigned short u; } cv;
  cv.b = (__bf16)f;
  return cv.u;
}

// ---------------- degree / norm ----------------
__global__ void deg_kernel(const int* __restrict__ edst, float* __restrict__ deg, int E) {
  int i = blockIdx.x * blockDim.x + threadIdx.x;
  if (i < E) atomicAdd(&deg[edst[i]], 1.0f);
}

__global__ void norm_kernel(float* __restrict__ norm, int N) {
  int i = blockIdx.x * blockDim.x + threadIdx.x;
  if (i < N) { float d = norm[i]; norm[i] = rsqrtf(d < 1.0f ? 1.0f : d); }
}

// ---------------- xs = x * norm (row-scaled features) ----------------
__global__ void scale_kernel(const float* __restrict__ x, const float* __restrict__ norm,
                             float* __restrict__ xs, int N) {
  int i = blockIdx.x * blockDim.x + threadIdx.x;   // N*32 threads, float4 each
  if (i >= N * 32) return;
  int row = i >> 5, c = (i & 31) << 2;
  float nrm = norm[row];
  float4 v = *(const float4*)(x + (size_t)row * D_IN + c);
  float4 o; o.x = v.x * nrm; o.y = v.y * nrm; o.z = v.z * nrm; o.w = v.w * nrm;
  *(float4*)(xs + (size_t)row * D_IN + c) = o;
}

// ---------------- edge scatter-add: agg[dst] += xs[src] ----------------
__global__ void scatter_kernel(const int* __restrict__ esrc, const int* __restrict__ edst,
                               const float* __restrict__ xs, float* __restrict__ agg, int E) {
  int t = blockIdx.x * blockDim.x + threadIdx.x;   // one wave per edge
  int e = t >> 5;
  if (e >= E) return;
  int c = (t & 31) << 2;
  int s = esrc[e], d = edst[e];
  float4 v = *(const float4*)(xs + (size_t)s * D_IN + c);
  float* a = agg + (size_t)d * D_IN + c;
  atomicAdd(a + 0, v.x); atomicAdd(a + 1, v.y);
  atomicAdd(a + 2, v.z); atomicAdd(a + 3, v.w);
}

// ---------------- weights: transpose + bf16 convert ----------------
__global__ void prep_weights(const float* __restrict__ W1, const float* __restrict__ W2,
                             unsigned short* __restrict__ W1T, unsigned short* __restrict__ W2T) {
  int idx = blockIdx.x * blockDim.x + threadIdx.x;
  if (idx < D_IN * D_HID) {            // W1T[n][k] = bf16(W1[k][n])
    int n = idx >> 7, k = idx & 127;
    W1T[idx] = f2bf_bits(W1[(size_t)k * D_HID + n]);
  }
  if (idx < D_OUT * D_HID) {           // W2T[n][k] = bf16(W2[k][n])
    int n = idx >> 7, k = idx & 127;
    W2T[idx] = f2bf_bits(W2[(size_t)k * D_OUT + n]);
  }
}

// Build the 16-element bf16 A fragment from two 8-float runs, scaled by nrm.
static __device__ __forceinline__ v16bf make_a(const float* __restrict__ ap, float nrm) {
  float4 a0 = *(const float4*)(ap);
  float4 a1 = *(const float4*)(ap + 4);
  float4 a2 = *(const float4*)(ap + 16);
  float4 a3 = *(const float4*)(ap + 20);
  v16bf A;
  A[0]  = f2bf(a0.x * nrm); A[1]  = f2bf(a0.y * nrm); A[2]  = f2bf(a0.z * nrm); A[3]  = f2bf(a0.w * nrm);
  A[4]  = f2bf(a1.x * nrm); A[5]  = f2bf(a1.y * nrm); A[6]  = f2bf(a1.z * nrm); A[7]  = f2bf(a1.w * nrm);
  A[8]  = f2bf(a2.x * nrm); A[9]  = f2bf(a2.y * nrm); A[10] = f2bf(a2.z * nrm); A[11] = f2bf(a2.w * nrm);
  A[12] = f2bf(a3.x * nrm); A[13] = f2bf(a3.y * nrm); A[14] = f2bf(a3.z * nrm); A[15] = f2bf(a3.w * nrm);
  return A;
}

static __device__ __forceinline__ v16bf load_b(const unsigned short* __restrict__ p) {
  union { v16bf v; uint4 q[2]; } B;
  const uint4* bp = (const uint4*)p;
  B.q[0] = bp[0]; B.q[1] = bp[2];
  return B.v;
}

// ---------------- layer1: hs = ELU((norm.agg)@W1 + b1) * norm ----------------
__global__ __launch_bounds__(256) void layer1_kernel(
    const float* __restrict__ agg, const float* __restrict__ norm,
    const unsigned short* __restrict__ W1T, const float* __restrict__ b1,
    float* __restrict__ hs, int N) {
  const int lane = threadIdx.x & 31;
  const int wave = threadIdx.x >> 5;        // 8 waves -> 8 N-tiles of 16 (D_HID=128)
  const int half = lane >> 4;
  const int l16  = lane & 15;
  const int node0 = blockIdx.x * 16;
  int arow = node0 + l16; if (arow >= N) arow = N - 1;   // keep EXEC all-ones
  const int n = wave * 16 + l16;
  const float anrm = norm[arow];

  v8f acc = {};
  #pragma unroll
  for (int kk = 0; kk < D_IN; kk += 32) {
    // A (16x32 bf16): lane holds row M=l16; K = half*8 + {0..7} and 16 + half*8 + {0..7}
    v16bf A = make_a(agg + (size_t)arow * D_IN + kk + half * 8, anrm);
    // B (32x16 bf16): lane holds column N=l16; K = half*8 + {0..7} and 16 + half*8 + {0..7}
    v16bf B = load_b(W1T + (size_t)n * D_IN + kk + half * 8);
    acc = __builtin_amdgcn_wmma_f32_16x16x32_bf16(false, A, false, B,
                                                  (short)0, acc, false, false);
  }
  const float bias = b1[n];
  #pragma unroll
  for (int v = 0; v < 8; ++v) {            // D: VGPR v, lane -> M = half*8+v, N = n
    int row = node0 + half * 8 + v;
    if (row < N) {
      float x = acc[v] + bias;
      x = x > 0.0f ? x : (__expf(x) - 1.0f);          // ELU(alpha=1)
      hs[(size_t)row * D_HID + n] = x * norm[row];    // pre-scale for next prop
    }
  }
}

// ---------------- layer2: out = (norm.agg)@W2 + b2 ----------------
__global__ __launch_bounds__(128) void layer2_kernel(
    const float* __restrict__ agg, const float* __restrict__ norm,
    const unsigned short* __restrict__ W2T, const float* __restrict__ b2,
    float* __restrict__ out, int N) {
  const int lane = threadIdx.x & 31;
  const int wave = threadIdx.x >> 5;        // 4 waves -> 4 N-tiles of 16 (D_OUT=64)
  const int half = lane >> 4;
  const int l16  = lane & 15;
  const int node0 = blockIdx.x * 16;
  int arow = node0 + l16; if (arow >= N) arow = N - 1;
  const int n = wave * 16 + l16;
  const float anrm = norm[arow];

  v8f acc = {};
  #pragma unroll
  for (int kk = 0; kk < D_HID; kk += 32) {
    v16bf A = make_a(agg + (size_t)arow * D_HID + kk + half * 8, anrm);
    v16bf B = load_b(W2T + (size_t)n * D_HID + kk + half * 8);
    acc = __builtin_amdgcn_wmma_f32_16x16x32_bf16(false, A, false, B,
                                                  (short)0, acc, false, false);
  }
  const float bias = b2[n];
  #pragma unroll
  for (int v = 0; v < 8; ++v) {
    int row = node0 + half * 8 + v;
    if (row < N) out[(size_t)row * D_OUT + n] = acc[v] + bias;
  }
}

extern "C" void kernel_launch(void* const* d_in, const int* in_sizes, int n_in,
                              void* d_out, int out_size, void* d_ws, size_t ws_size,
                              hipStream_t stream) {
  const float* features = (const float*)d_in[0];
  const float* W1       = (const float*)d_in[1];
  const float* b1       = (const float*)d_in[2];
  const float* W2       = (const float*)d_in[3];
  const float* b2       = (const float*)d_in[4];
  const int*   esrc     = (const int*)d_in[5];
  const int*   edst     = (const int*)d_in[6];
  float*       out      = (float*)d_out;

  const int N = in_sizes[0] / D_IN;
  const int E = in_sizes[5];

  // workspace carve-up (256B aligned)
  char* ws = (char*)d_ws;
  size_t off = 0;
  auto carve = [&](size_t bytes) -> char* {
    char* p = ws + off;
    off = (off + bytes + 255) & ~(size_t)255;
    return p;
  };
  float*          norm = (float*)carve((size_t)N * sizeof(float));
  float*          xs   = (float*)carve((size_t)N * D_IN * sizeof(float));   // reused as hs
  float*          agg  = (float*)carve((size_t)N * D_IN * sizeof(float));
  unsigned short* W1T  = (unsigned short*)carve((size_t)D_HID * D_IN * sizeof(unsigned short));
  unsigned short* W2T  = (unsigned short*)carve((size_t)D_OUT * D_HID * sizeof(unsigned short));
  (void)n_in; (void)out_size; (void)ws_size;

  const int TB = 256;
  const int mblocks = (N + 15) / 16;
  const long long sthreads = (long long)E * 32;

  // degree + norm
  hipMemsetAsync(norm, 0, (size_t)N * sizeof(float), stream);
  deg_kernel<<<(E + TB - 1) / TB, TB, 0, stream>>>(edst, norm, E);
  norm_kernel<<<(N + TB - 1) / TB, TB, 0, stream>>>(norm, N);
  // weights (independent of graph work, just ordered on stream)
  prep_weights<<<(D_IN * D_HID + TB - 1) / TB, TB, 0, stream>>>(W1, W2, W1T, W2T);

  // prop 1
  scale_kernel<<<((N * 32) + TB - 1) / TB, TB, 0, stream>>>(features, norm, xs, N);
  hipMemsetAsync(agg, 0, (size_t)N * D_IN * sizeof(float), stream);
  scatter_kernel<<<(unsigned)((sthreads + TB - 1) / TB), TB, 0, stream>>>(esrc, edst, xs, agg, E);
  // layer 1 (WMMA) -> hs stored pre-scaled by norm (reuses xs buffer)
  layer1_kernel<<<mblocks, 256, 0, stream>>>(agg, norm, W1T, b1, xs, N);

  // prop 2
  hipMemsetAsync(agg, 0, (size_t)N * D_HID * sizeof(float), stream);
  scatter_kernel<<<(unsigned)((sthreads + TB - 1) / TB), TB, 0, stream>>>(esrc, edst, xs, agg, E);
  // layer 2 (WMMA) -> d_out
  layer2_kernel<<<mblocks, 128, 0, stream>>>(agg, norm, W2T, b2, out, N);
}
